// EncoderLSTMStack_46608985096547
// MI455X (gfx1250) — matrix operand order repaired and checked by reading, not compiled, not run on hardware
//
#include <hip/hip_runtime.h>
#include <hip/hip_bf16.h>
#include <stdint.h>

#define T_    512
#define E_    1024
#define H_    1024
#define K3_   3072
#define G4_   4096
#define OUT_  256
#define NWG_  32
#define NCELL_ 4

typedef __bf16 bf16_t;
typedef __attribute__((ext_vector_type(16))) __bf16 v16bf;
typedef __attribute__((ext_vector_type(8)))  __bf16 v8bf;
typedef __attribute__((ext_vector_type(8)))  float  v8f;
typedef __attribute__((ext_vector_type(4)))  int    v4i;

#if __has_builtin(__builtin_amdgcn_global_load_async_to_lds_b128)
#define ASYNC_LDS 1
typedef __attribute__((address_space(1))) v4i*  gv4i_p;
typedef __attribute__((address_space(3))) v4i*  lv4i_p;
typedef __attribute__((address_space(1))) char* gchar_p;
typedef __attribute__((address_space(3))) char* lchar_p;
#endif

// ---------------- workspace layout (bytes) ----------------
static constexpr size_t SZ_WG   = (size_t)G4_ * K3_ * 2;       // 24 MB bf16 fused [W_ih|W_hh]
static constexpr size_t OFF_WGT = 0;
static constexpr size_t SZ_W1   = (size_t)H_ * H_ * 2;         // 2 MB bf16 square mats
static constexpr size_t OFF_SQ  = OFF_WGT + 4 * SZ_WG;         // Wout,Wv,Wo per cell
static constexpr size_t OFF_BG  = OFF_SQ + 4 * 3 * SZ_W1;      // fused gate bias (f32)
static constexpr size_t SZ_BG   = (size_t)G4_ * 4;
static constexpr size_t OFF_ST  = OFF_BG + 4 * SZ_BG;          // per-cell state block
static constexpr size_t ST_H = 0, ST_C = 4096, ST_R = 8192, ST_GATES = 12288,
                         ST_RO = 28672, ST_RO16 = 32768, ST_HL16 = 34816,
                         ST_XCAT = 36864, ST_S = 43008, ST_RSC = 45056;
static constexpr size_t SZ_ST   = 47360;
static constexpr size_t OFF_V   = OFF_ST + 4 * SZ_ST;          // V matrices (T x H f32)
static constexpr size_t SZ_V    = (size_t)T_ * H_ * 4;
static constexpr size_t OFF_L0  = OFF_V + 4 * SZ_V;            // layer0 outputs per dir
static constexpr size_t OFF_MEM = OFF_L0 + 2 * SZ_V;           // layer1 outputs (time-aligned)
static constexpr size_t OFF_SY  = OFF_MEM + 2 * SZ_V;          // sync block

struct SyncBlk { unsigned cnt[NCELL_]; unsigned gen[NCELL_]; unsigned progress[NCELL_]; };

struct CellIn {
  const float *W_ih, *b_ih, *W_hh, *b_hh, *W_out, *b_out,
              *W_d, *b_d, *W_u, *b_u, *W_v, *b_v, *W_o, *b_o;
};
struct PArgs { const float* embeds; CellIn cell[NCELL_]; };
struct ConvArgs {
  const float *W_ih[4], *b_ih[4], *W_hh[4], *b_hh[4], *W_out[4], *W_v[4], *W_o[4];
};

// ---------------- helpers ----------------
__device__ inline void cell_barrier(unsigned* cnt, unsigned* gen) {
  __syncthreads();
  if (threadIdx.x == 0) {
    __threadfence();
    unsigned g = __hip_atomic_load(gen, __ATOMIC_RELAXED, __HIP_MEMORY_SCOPE_AGENT);
    unsigned arrived = __hip_atomic_fetch_add(cnt, 1u, __ATOMIC_ACQ_REL,
                                              __HIP_MEMORY_SCOPE_AGENT) + 1u;
    if (arrived == NWG_) {
      __hip_atomic_store(cnt, 0u, __ATOMIC_RELAXED, __HIP_MEMORY_SCOPE_AGENT);
      __hip_atomic_fetch_add(gen, 1u, __ATOMIC_ACQ_REL, __HIP_MEMORY_SCOPE_AGENT);
    } else {
      while (__hip_atomic_load(gen, __ATOMIC_ACQUIRE, __HIP_MEMORY_SCOPE_AGENT) == g)
        __builtin_amdgcn_s_sleep(2);
    }
    __threadfence();
  }
  __syncthreads();
}

// 16-row GEMV slice via v_wmma_f32_16x16x32_bf16.
// A tile (16x32): lanes 0-15 carry K {0..15}, lanes 16-31 carry {16..31} (matches B's K order).
// B tile (32x16): each lane holds 16 contiguous K of x (half selected by lane>=16),
// replicating x into all 16 columns -> every C column equals W.x for these 16 rows.
__device__ inline v8f gemv_wmma16(const bf16_t* __restrict__ W, int ldk, int row_base,
                                  const bf16_t* sVec, int Klen) {
  const int lane   = threadIdx.x & 31;
  const int laneHi = lane >> 4;
  const int row    = row_base + (lane & 15);
  const bf16_t* wp = W + (size_t)row * ldk + (laneHi ? 16 : 0);
  const bf16_t* xp = sVec + (laneHi ? 16 : 0);
  v8f acc = {};
  for (int k0 = 0; k0 < Klen; k0 += 32) {
    union { v16bf v; v8bf h[2]; } ua;
    ua.h[0] = *(const v8bf*)(wp + k0);
    ua.h[1] = *(const v8bf*)(wp + k0 + 8);
    v16bf b = *(const v16bf*)(xp + k0);
    __builtin_prefetch(wp + k0 + 256, 0, 0);   // -> global_prefetch_b8 on weight stream
    acc = __builtin_amdgcn_wmma_f32_16x16x32_bf16(false, ua.v, false, b,
                                                  (short)0, acc, false, false);
  }
  return acc;
}

// ---------------- weight conversion (fp32 -> bf16, fused layouts) ----------------
__global__ __launch_bounds__(256) void convert_gate_weights(ConvArgs a, char* ws) {
  int bid = blockIdx.x;                 // 4 cells * 4096 rows
  int c = bid >> 12, r = bid & 4095;
  bf16_t* dst = (bf16_t*)(ws + OFF_WGT + (size_t)c * SZ_WG) + (size_t)r * K3_;
  const float* wih = a.W_ih[c] + (size_t)r * 2048;
  const float* whh = a.W_hh[c] + (size_t)r * 1024;
  for (int k = threadIdx.x; k < K3_; k += 256)
    dst[k] = (bf16_t)(k < 2048 ? wih[k] : whh[k - 2048]);
  if (threadIdx.x == 0) {
    float* bg = (float*)(ws + OFF_BG + (size_t)c * SZ_BG);
    bg[r] = a.b_ih[c][r] + a.b_hh[c][r];
  }
}

__global__ __launch_bounds__(256) void convert_sq_weights(ConvArgs a, char* ws) {
  int bid = blockIdx.x;                 // 4 cells * 3 mats * 1024 rows
  int c = bid / 3072, rem = bid % 3072, m = rem >> 10, r = rem & 1023;
  const float* src = (m == 0 ? a.W_out[c] : (m == 1 ? a.W_v[c] : a.W_o[c])) + (size_t)r * H_;
  bf16_t* dst = (bf16_t*)(ws + OFF_SQ + (size_t)(c * 3 + m) * SZ_W1) + (size_t)r * H_;
  for (int k = threadIdx.x; k < H_; k += 256) dst[k] = (bf16_t)src[k];
}

// ---------------- persistent LSTM-stack kernel ----------------
__global__ __launch_bounds__(256, 1) void lstm_persistent(PArgs a, char* ws) {
  const int tid  = threadIdx.x;
  const int wave = tid >> 5;
  const int lane = tid & 31;
  const int cell = blockIdx.x / NWG_;   // 0=f0 1=f1 2=b0 3=b1
  const int wg   = blockIdx.x % NWG_;
  const int layer = cell & 1;
  const int dir   = cell >> 1;

  __shared__ __align__(32) bf16_t sX[K3_];   // broadcast input vector (bf16)
  __shared__ float sRed[256];
  __shared__ float sRsc[T_];
  __shared__ float sDU[2];

  const bf16_t* Wg   = (const bf16_t*)(ws + OFF_WGT + (size_t)cell * SZ_WG);
  const bf16_t* Wout = (const bf16_t*)(ws + OFF_SQ + (size_t)(cell * 3 + 0) * SZ_W1);
  const bf16_t* Wv   = (const bf16_t*)(ws + OFF_SQ + (size_t)(cell * 3 + 1) * SZ_W1);
  const bf16_t* Wo   = (const bf16_t*)(ws + OFF_SQ + (size_t)(cell * 3 + 2) * SZ_W1);
  const float*  bg   = (const float*)(ws + OFF_BG + (size_t)cell * SZ_BG);
  char* st = ws + OFF_ST + (size_t)cell * SZ_ST;
  float*  c_g     = (float*)(st + ST_C);
  float*  r_g     = (float*)(st + ST_R);
  float*  gates_g = (float*)(st + ST_GATES);
  float*  ro_g    = (float*)(st + ST_RO);
  bf16_t* ro16_g  = (bf16_t*)(st + ST_RO16);
  bf16_t* hl16_g  = (bf16_t*)(st + ST_HL16);
  bf16_t* xcat_g  = (bf16_t*)(st + ST_XCAT);     // [x(1024), r(1024), h(1024)] bf16
  float*  s_g     = (float*)(st + ST_S);
  float*  rsc_g   = (float*)(st + ST_RSC);
  float*  V_g     = (float*)(ws + OFF_V  + (size_t)cell * SZ_V);
  float*  out0_g  = (float*)(ws + OFF_L0 + (size_t)dir  * SZ_V);
  float*  mem_g   = (float*)(ws + OFF_MEM + (size_t)dir * SZ_V);
  SyncBlk* sy = (SyncBlk*)(ws + OFF_SY);
  unsigned* cnt = &sy->cnt[cell];
  unsigned* gen = &sy->gen[cell];
  const CellIn ci = a.cell[cell];

  for (int t = 0; t < T_; ++t) {
    // ---- S0: build x-part of fused input vector (layer1 waits on layer0 progress) ----
    if (wg < 4) {
      if (layer == 1) {
        if (tid == 0) {
          while (__hip_atomic_load(&sy->progress[cell - 1], __ATOMIC_ACQUIRE,
                                   __HIP_MEMORY_SCOPE_AGENT) <= (unsigned)t)
            __builtin_amdgcn_s_sleep(2);
          __threadfence();
        }
        __syncthreads();
      }
      int idx = wg * 256 + tid;
      float xv = layer ? out0_g[(size_t)t * H_ + idx]
                       : a.embeds[(size_t)(dir ? (T_ - 1 - t) : t) * E_ + idx];
      xcat_g[idx] = (bf16_t)xv;
    }
    cell_barrier(cnt, gen);

    // ---- A: gates = [W_ih|W_hh] . [x,r,h] + (b_ih+b_hh), 128 rows/WG via WMMA ----
    {
#if defined(ASYNC_LDS)
      for (int s4 = tid; s4 < 384; s4 += 256)
        __builtin_amdgcn_global_load_async_to_lds_b128(
            (gv4i_p)((gchar_p)(char*)xcat_g + s4 * 16),
            (lv4i_p)((lchar_p)(char*)sX + s4 * 16), 0, 0);
#if __has_builtin(__builtin_amdgcn_s_wait_asynccnt)
      __builtin_amdgcn_s_wait_asynccnt(0);
#else
      asm volatile("s_wait_asynccnt 0" ::: "memory");
#endif
#else
      uint4* dl = (uint4*)sX;
      const uint4* sg = (const uint4*)xcat_g;
      for (int i = tid; i < 384; i += 256) dl[i] = sg[i];
#endif
    }
    __syncthreads();
    {
      int row_base = wg * 128 + wave * 16;
      v8f acc = gemv_wmma16(Wg, K3_, row_base, sX, K3_);
      if (lane == 0 || lane == 16) {
        int r0 = row_base + (lane == 16 ? 8 : 0);
#pragma unroll
        for (int j = 0; j < 8; ++j) gates_g[r0 + j] = acc[j] + bg[r0 + j];
      }
    }
    cell_barrier(cnt, gen);

    // ---- B: LSTM pointwise, 32 dims/WG ----
    if (tid < 32) {
      int dim = wg * 32 + tid;
      float ig = gates_g[dim], fg = gates_g[H_ + dim],
            gg = gates_g[2 * H_ + dim], og = gates_g[3 * H_ + dim];
      float cn = (1.f / (1.f + __expf(-fg))) * c_g[dim] +
                 (1.f / (1.f + __expf(-ig))) * tanhf(gg);
      float hl = (1.f / (1.f + __expf(-og))) * tanhf(cn);
      c_g[dim] = cn;
      hl16_g[dim] = (bf16_t)hl;
    }
    cell_barrier(cnt, gen);

    // ---- C: rnn_output = W_out . h_lstm + b_out ----
    if (tid < 128) ((uint4*)sX)[tid] = ((const uint4*)hl16_g)[tid];
    __syncthreads();
    if (wave < 2) {
      int row_base = wg * 32 + wave * 16;
      v8f acc = gemv_wmma16(Wout, H_, row_base, sX, H_);
      if (lane == 0 || lane == 16) {
        int r0 = row_base + (lane == 16 ? 8 : 0);
#pragma unroll
        for (int j = 0; j < 8; ++j) {
          float v = acc[j] + ci.b_out[r0 + j];
          ro_g[r0 + j] = v;
          ro16_g[r0 + j] = (bf16_t)v;
          xcat_g[2 * H_ + r0 + j] = (bf16_t)v;   // recurrent h for next step
        }
      }
    }
    cell_barrier(cnt, gen);

    // ---- D: v_t = W_v.ro + b_v -> V[t];  o_t = W_o.ro + b_o -> stack output;  d,u dots ----
    if (tid < 128) ((uint4*)sX)[tid] = ((const uint4*)ro16_g)[tid];
    __syncthreads();
    if (wave < 2) {
      int row_base = wg * 32 + wave * 16;
      v8f acc = gemv_wmma16(Wv, H_, row_base, sX, H_);
      if (lane == 0 || lane == 16) {
        int r0 = row_base + (lane == 16 ? 8 : 0);
#pragma unroll
        for (int j = 0; j < 8; ++j) V_g[(size_t)t * H_ + r0 + j] = acc[j] + ci.b_v[r0 + j];
      }
    } else if (wave < 4) {
      int row_base = wg * 32 + (wave - 2) * 16;
      v8f acc = gemv_wmma16(Wo, H_, row_base, sX, H_);
      if (lane == 0 || lane == 16) {
        int r0 = row_base + (lane == 16 ? 8 : 0);
        float* dst = layer ? (mem_g + (size_t)(dir ? (T_ - 1 - t) : t) * H_)
                           : (out0_g + (size_t)t * H_);
#pragma unroll
        for (int j = 0; j < 8; ++j) dst[r0 + j] = acc[j] + ci.b_o[r0 + j];
      }
    }
    __syncthreads();
    if (wg == 0) {
      if (tid >= 128) {
        int k = tid - 128;
        float pd = 0.f, pu = 0.f;
        for (int kk = k; kk < H_; kk += 128) {
          float rv = ro_g[kk];
          pd += ci.W_d[kk] * rv;
          pu += ci.W_u[kk] * rv;
        }
        sRed[k] = pd; sRsc[k] = pu;
      }
      __syncthreads();
      if (tid == 0) {
        float sd = 0.f, su = 0.f;
        for (int k = 0; k < 128; ++k) { sd += sRed[k]; su += sRsc[k]; }
        sDU[0] = sd + ci.b_d[0];
        sDU[1] = su + ci.b_u[0];
      }
    }
    cell_barrier(cnt, gen);

    // ---- E: stick-breaking memory scalars (WG0/thread0; suffix scans bounded by t) ----
    if (wg == 0 && tid == 0) {
      __threadfence();
      __hip_atomic_store(&sy->progress[cell], (unsigned)(t + 1),
                         __ATOMIC_RELEASE, __HIP_MEMORY_SCOPE_AGENT);
      float d = sDU[0], u = sDU[1];
      float run = 0.f;
      for (int i = t; i >= 0; --i) {           // suffix_prev over old s, then s update
        float si = s_g[i];
        float ns = fmaxf(si - fmaxf(u - run, 0.f), 0.f);
        run += si;
        s_g[i] = ns;
      }
      s_g[t] = d;
      run = 0.f;
      for (int i = t; i >= 0; --i) {           // suffix_new over new s, then r_sc
        float si = s_g[i];
        rsc_g[i] = fminf(si, fmaxf(1.f - run, 0.f));
        run += si;
      }
    }
    cell_barrier(cnt, gen);

    // ---- F: r = r_sc @ V (only rows 0..t nonzero), 32 dims/WG ----
    for (int i = tid; i < T_; i += 256) sRsc[i] = rsc_g[i];
    __syncthreads();
    {
      int dim = wg * 32 + (tid & 31);
      int chunk = tid >> 5;
      float p = 0.f;
      for (int tp = chunk; tp <= t; tp += 8)
        p += sRsc[tp] * V_g[(size_t)tp * H_ + dim];
      sRed[tid] = p;
    }
    __syncthreads();
    if (tid < 32) {
      float rv = 0.f;
#pragma unroll
      for (int j = 0; j < 8; ++j) rv += sRed[j * 32 + tid];
      int dim = wg * 32 + tid;
      r_g[dim] = rv;
      xcat_g[H_ + dim] = (bf16_t)rv;           // recurrent r for next step
    }
    cell_barrier(cnt, gen);
  }
}

// ---------------- attention + MLP head (single workgroup) ----------------
__global__ __launch_bounds__(256) void attn_head_kernel(
    const float* q, const float* attn_w, const float* W1, const float* b1,
    const float* W2, const float* b2, const char* ws, float* out) {
  __shared__ float sW[2 * H_];
  __shared__ float sL[T_];
  __shared__ float sC[2 * H_];
  __shared__ float sH1[32];
  const float* memf = (const float*)(ws + OFF_MEM);
  const float* memb = (const float*)(ws + OFF_MEM + SZ_V);
  int tid = threadIdx.x;
  for (int j = tid; j < 2 * H_; j += 256) {
    float acc = 0.f;
    for (int k = 0; k < 64; ++k) acc += q[k] * attn_w[(size_t)k * 2 * H_ + j];
    sW[j] = acc;
  }
  __syncthreads();
  for (int t = tid; t < T_; t += 256) {
    float acc = 0.f;
    const float* ef = memf + (size_t)t * H_;
    const float* eb = memb + (size_t)t * H_;
    for (int j = 0; j < H_; ++j) acc += sW[j] * ef[j];
    for (int j = 0; j < H_; ++j) acc += sW[H_ + j] * eb[j];
    sL[t] = acc;
  }
  __syncthreads();
  if (tid == 0) {
    float m = -1e30f;
    for (int t = 0; t < T_; ++t) m = fmaxf(m, sL[t]);
    float ssum = 0.f;
    for (int t = 0; t < T_; ++t) { float e = __expf(sL[t] - m); sL[t] = e; ssum += e; }
    float inv = 1.f / ssum;
    for (int t = 0; t < T_; ++t) sL[t] *= inv;
  }
  __syncthreads();
  for (int j = tid; j < 2 * H_; j += 256) {
    float acc = 0.f;
    const float* col = (j < H_) ? (memf + j) : (memb + (j - H_));
    for (int t = 0; t < T_; ++t) acc += sL[t] * col[(size_t)t * H_];
    sC[j] = acc;
  }
  __syncthreads();
  if (tid < 32) {
    float acc = b1[tid];
    const float* w = W1 + (size_t)tid * 2 * H_;
    for (int j = 0; j < 2 * H_; ++j) acc += w[j] * sC[j];
    sH1[tid] = fmaxf(acc, 0.f);
  }
  __syncthreads();
  if (tid < OUT_) {
    float acc = b2[tid];
    const float* w = W2 + (size_t)tid * 32;
    for (int i = 0; i < 32; ++i) acc += w[i] * sH1[i];
    out[tid] = 1.f / (1.f + __expf(-acc));
  }
}

// ---------------- host launch ----------------
extern "C" void kernel_launch(void* const* d_in, const int* in_sizes, int n_in,
                              void* d_out, int out_size, void* d_ws, size_t ws_size,
                              hipStream_t stream) {
  (void)in_sizes; (void)n_in; (void)out_size; (void)ws_size;
  char* ws = (char*)d_ws;
  PArgs pa; ConvArgs ca;
  pa.embeds = (const float*)d_in[0];
  static const int base[4] = {1, 15, 29, 43};   // f0, f1, b0, b1
  for (int c = 0; c < 4; ++c) {
    const int b = base[c];
    pa.cell[c].W_ih = (const float*)d_in[b + 0];  pa.cell[c].b_ih = (const float*)d_in[b + 1];
    pa.cell[c].W_hh = (const float*)d_in[b + 2];  pa.cell[c].b_hh = (const float*)d_in[b + 3];
    pa.cell[c].W_out= (const float*)d_in[b + 4];  pa.cell[c].b_out= (const float*)d_in[b + 5];
    pa.cell[c].W_d  = (const float*)d_in[b + 6];  pa.cell[c].b_d  = (const float*)d_in[b + 7];
    pa.cell[c].W_u  = (const float*)d_in[b + 8];  pa.cell[c].b_u  = (const float*)d_in[b + 9];
    pa.cell[c].W_v  = (const float*)d_in[b + 10]; pa.cell[c].b_v  = (const float*)d_in[b + 11];
    pa.cell[c].W_o  = (const float*)d_in[b + 12]; pa.cell[c].b_o  = (const float*)d_in[b + 13];
    ca.W_ih[c] = pa.cell[c].W_ih; ca.b_ih[c] = pa.cell[c].b_ih;
    ca.W_hh[c] = pa.cell[c].W_hh; ca.b_hh[c] = pa.cell[c].b_hh;
    ca.W_out[c] = pa.cell[c].W_out; ca.W_v[c] = pa.cell[c].W_v; ca.W_o[c] = pa.cell[c].W_o;
  }
  // zero per-cell state (h,c,r,s,rsc,xcat) and sync counters each call (graph-replay safe)
  (void)hipMemsetAsync(ws + OFF_ST, 0, 4 * SZ_ST, stream);
  (void)hipMemsetAsync(ws + OFF_SY, 0, sizeof(SyncBlk), stream);

  convert_gate_weights<<<4 * G4_, 256, 0, stream>>>(ca, ws);
  convert_sq_weights<<<4 * 3 * 1024, 256, 0, stream>>>(ca, ws);
  lstm_persistent<<<NCELL_ * NWG_, 256, 0, stream>>>(pa, ws);
  attn_head_kernel<<<1, 256, 0, stream>>>(
      (const float*)d_in[57], (const float*)d_in[58], (const float*)d_in[59],
      (const float*)d_in[60], (const float*)d_in[61], (const float*)d_in[62],
      ws, (float*)d_out);
}
